// AveragedKeyCircularConvolutionalAttention_47983374631613
// MI455X (gfx1250) — compile-verified, matrix operand order checked by reading, not run on hardware
//
#include <hip/hip_runtime.h>
#include <hip/hip_bf16.h>

// ---------------------------------------------------------------------------
// AveragedKeyCircularConvolutionalAttention for MI455X (gfx1250, wave32, WMMA)
// B=2, N=2048, C=768, H=12, D=64, SCALE = 1/8
// GEMMs feed v_wmma_f32_16x16x32_bf16 from LDS tiles staged by the Tensor
// Data Mover (tensor_load_to_lds, TENSORcnt), double-buffered.
// ---------------------------------------------------------------------------

#define BQ 2
#define SEQ 2048
#define CH 768
#define NH 12
#define HD 64
#define ROWS (BQ * SEQ)   // 4096

typedef __attribute__((ext_vector_type(16))) __bf16 v16bf;
typedef __attribute__((ext_vector_type(8)))  float  v8f;
typedef __attribute__((ext_vector_type(4)))  unsigned int u32x4;
typedef __attribute__((ext_vector_type(8)))  int         i32x8;
typedef __attribute__((ext_vector_type(4)))  int         i32x4;

// LDS tile geometry for TDM-staged GEMM operands: 128 rows x 32 bf16 (64B),
// padded by the TDM to an 80B row stride (16 DWORD interval, 4 DWORD pad)
// so b128 fragment reads are LDS-bank-conflict free (row*20 mod 64 is a perm).
#define TILE_ROW_BYTES 80
#define TILE_BYTES (128 * TILE_ROW_BYTES)   // 10240

__device__ __forceinline__ unsigned lds_off(const void* p) {
    // Generic LDS pointer low 32 bits == byte offset within workgroup LDS.
    return (unsigned)(unsigned long long)p;
}

// Issue one TDM 2D tile load: 32 (k) x 128 (rows) bf16 elements from a
// row-major matrix with row length `stride0` elements, into LDS at lds_addr,
// with pad-on-load producing an 80-byte LDS row stride.
__device__ __forceinline__ void tdm_load_tile(unsigned lds_addr,
                                              const void* gaddr,
                                              unsigned tdim0, unsigned tdim1,
                                              unsigned stride0) {
    unsigned long long ga = (unsigned long long)gaddr;
    u32x4 g0;
    g0[0] = 1u;                                   // count=1, user descriptor
    g0[1] = lds_addr;                             // lds_addr (bytes)
    g0[2] = (unsigned)(ga & 0xffffffffull);       // global_addr[31:0]
    g0[3] = (unsigned)((ga >> 32) & 0x01ffffffull)// global_addr[56:32]
            | 0x80000000u;                        // type=2 ("image")
    i32x8 g1;
    g1[0] = 0x00010000                            // data_size=1 -> 2 bytes
          | (1 << 20)                             // pad_enable
          | (3 << 22)                             // pad_interval: 16 DWORDs
          | (3 << 25);                            // pad_amount:    4 DWORDs
    g1[1] = (int)((tdim0 & 0xffffu) << 16);       // tensor_dim0[15:0]
    g1[2] = (int)((tdim0 >> 16) | ((tdim1 & 0xffffu) << 16)); // dim0 hi/dim1 lo
    g1[3] = (int)((tdim1 >> 16) | (32u << 16));   // tile_dim0 = 32 elements
    g1[4] = (int)128;                             // tile_dim1 = 128, tile_dim2=0
    g1[5] = (int)stride0;                         // tensor_dim0_stride (elems)
    g1[6] = 0;
    g1[7] = 0;
    i32x4 zz4 = {0, 0, 0, 0};                     // groups 2/3: 2D tensor
    i32x8 zz8 = {0, 0, 0, 0, 0, 0, 0, 0};         // 6-arg toolchain extra group
    __builtin_amdgcn_tensor_load_to_lds(g0, g1, zz4, zz4, zz8, 0);
}

// Load a 16x32 bf16 WMMA fragment from an LDS tile row pointer that already
// includes (+ half8 elements). Chunks: K 0..7 / 16..23 (lanes 0-15) or
// K 8..15 / 24..31 (lanes 16-31), per the ISA 16-bit A/B layout.
__device__ __forceinline__ v16bf lds_frag(const char* p) {
    union { uint4 u[2]; v16bf v; } f;
    f.u[0] = *reinterpret_cast<const uint4*>(p);
    f.u[1] = *reinterpret_cast<const uint4*>(p + 32);
    return f.v;
}

__device__ __forceinline__ v16bf load_frag16(const __hip_bfloat16* p) {
    union { uint4 u[2]; v16bf v; } f;
    f.u[0] = *reinterpret_cast<const uint4*>(p);
    f.u[1] = *reinterpret_cast<const uint4*>(p + 16);
    return f.v;
}

// ---------------------------------------------------------------------------
// Kernel 0: f32 -> bf16 conversion
// ---------------------------------------------------------------------------
__global__ void cvt_bf16_kernel(const float* __restrict__ src,
                                __hip_bfloat16* __restrict__ dst, int n) {
    int i = blockIdx.x * blockDim.x + threadIdx.x;
    if (i < n) dst[i] = __float2bfloat16(src[i]);
}

// ---------------------------------------------------------------------------
// Kernel 1: GEMM  out[m,n] = sum_k A[m,k] * W[n,k]  via TDM-staged LDS tiles.
// MODE 0/1: store f32 (Q/K)   MODE 2: bf16 transposed per head -> Vt
// MODE 3: store f32 + bias (final projection -> d_out)
// Block: 256 threads (8 waves), 128x128 tile; wave = 32(M) x 64(N).
// ---------------------------------------------------------------------------
template <int MODE>
__global__ void gemm_bf16_kernel(const __hip_bfloat16* __restrict__ A,
                                 const __hip_bfloat16* __restrict__ W,
                                 float* __restrict__ outF,
                                 __hip_bfloat16* __restrict__ outVt,
                                 const float* __restrict__ bias) {
    __shared__ __align__(16) char smem[4 * TILE_BYTES];  // A0 B0 A1 B1

    const int n0 = blockIdx.x * 128;
    const int m0 = blockIdx.y * 128;
    const int w    = threadIdx.x >> 5;
    const int lane = threadIdx.x & 31;
    const int mwave = w >> 1;          // 0..3
    const int nwave = w & 1;           // 0..1
    const int r     = lane & 15;
    const int half8 = (lane >> 4) << 3;

    const unsigned smem0 = lds_off(smem);
    const bool issuer = (threadIdx.x < 32);

    // Prime the pipeline: K-step 0 into buffer 0.
    if (issuer) {
        tdm_load_tile(smem0 + 0 * TILE_BYTES, A + (size_t)m0 * CH, CH, ROWS, CH);
        tdm_load_tile(smem0 + 1 * TILE_BYTES, W + (size_t)n0 * CH, CH, CH, CH);
    }

    v8f zero8 = {0.f, 0.f, 0.f, 0.f, 0.f, 0.f, 0.f, 0.f};
    v8f acc[2][4];
#pragma unroll
    for (int mt = 0; mt < 2; ++mt)
#pragma unroll
        for (int nt = 0; nt < 4; ++nt) acc[mt][nt] = zero8;

    // Per-lane LDS fragment base offsets within a tile.
    const char* abase[2];
    const char* bbase[4];
#pragma unroll
    for (int mt = 0; mt < 2; ++mt)
        abase[mt] = smem + (mwave * 32 + mt * 16 + r) * TILE_ROW_BYTES + half8 * 2;
#pragma unroll
    for (int nt = 0; nt < 4; ++nt)
        bbase[nt] = smem + (nwave * 64 + nt * 16 + r) * TILE_ROW_BYTES + half8 * 2;

    for (int k0 = 0; k0 < CH; k0 += 32) {
        const int cur = (k0 >> 5) & 1;
        // Wait for the DMA that filled buffer `cur`, then publish to all waves.
        if (issuer) __builtin_amdgcn_s_wait_tensorcnt(0);
        __syncthreads();
        // Kick off the next K-step's tiles into the other buffer (overlaps
        // with the WMMA work below).
        if (issuer && (k0 + 32 < CH)) {
            tdm_load_tile(smem0 + (2 * (cur ^ 1) + 0) * TILE_BYTES,
                          A + (size_t)m0 * CH + (k0 + 32), CH, ROWS, CH);
            tdm_load_tile(smem0 + (2 * (cur ^ 1) + 1) * TILE_BYTES,
                          W + (size_t)n0 * CH + (k0 + 32), CH, CH, CH);
        }

        const int boff = 2 * cur * TILE_BYTES;
        v16bf af[2], bf[4];
#pragma unroll
        for (int mt = 0; mt < 2; ++mt) af[mt] = lds_frag(abase[mt] + boff);
#pragma unroll
        for (int nt = 0; nt < 4; ++nt)
            bf[nt] = lds_frag(bbase[nt] + boff + TILE_BYTES);
#pragma unroll
        for (int mt = 0; mt < 2; ++mt)
#pragma unroll
            for (int nt = 0; nt < 4; ++nt)
                acc[mt][nt] = __builtin_amdgcn_wmma_f32_16x16x32_bf16(
                    false, af[mt], false, bf[nt], (short)0, acc[mt][nt],
                    false, false);
    }

    // D layout: lane -> column (lane&15); VGPR v -> row v (+8 for lanes>=16)
#pragma unroll
    for (int mt = 0; mt < 2; ++mt) {
#pragma unroll
        for (int nt = 0; nt < 4; ++nt) {
            const int ncol  = n0 + nwave * 64 + nt * 16 + r;
            const int mbase = m0 + mwave * 32 + mt * 16 + half8;
            v8f a = acc[mt][nt];
            if (MODE == 2) {
                // transpose-per-head store: Vt[((b*12+h)*64+d)*2048 + j]
                __hip_bfloat16 tmp[8];
#pragma unroll
                for (int rr = 0; rr < 8; ++rr) tmp[rr] = __float2bfloat16(a[rr]);
                const int bb = mbase >> 11, j = mbase & 2047;
                const int h = ncol >> 6, dd = ncol & 63;
                *reinterpret_cast<uint4*>(
                    outVt + ((size_t)((bb * NH + h) * HD + dd)) * SEQ + j) =
                    *reinterpret_cast<uint4*>(tmp);
            } else {
                const float bv = (MODE == 3) ? bias[ncol] : 0.0f;
#pragma unroll
                for (int rr = 0; rr < 8; ++rr)
                    outF[(size_t)(mbase + rr) * CH + ncol] = a[rr] + bv;
            }
        }
    }
}

// ---------------------------------------------------------------------------
// Kernel 2: per (b,h): K_avg, z = scale * Q . K_avg, softmax, store bf16
// attn with the 1/N of the circulant matrix folded in.
// ---------------------------------------------------------------------------
__global__ void kavg_softmax_kernel(const float* __restrict__ Qf,
                                    const float* __restrict__ Kf,
                                    __hip_bfloat16* __restrict__ attnb) {
    __shared__ float s_kavg[HD];
    __shared__ float s_red[256];
    __shared__ float s_z[SEQ];

    const int bh = blockIdx.x;
    const int b = bh / NH, h = bh % NH;
    const int t = threadIdx.x;

    // mean of K over sequence
    {
        const int d = t & 63, seg = t >> 6;
        float s = 0.f;
        for (int j = seg; j < SEQ; j += 4)
            s += Kf[(size_t)(b * SEQ + j) * CH + h * HD + d];
        s_red[t] = s;
        __syncthreads();
        if (t < HD)
            s_kavg[t] = (s_red[t] + s_red[t + 64] + s_red[t + 128] +
                         s_red[t + 192]) * (1.0f / (float)SEQ);
        __syncthreads();
    }

    // z_i = scale * <Q_i, K_avg>
    float lmax = -1e30f;
    for (int i = t; i < SEQ; i += 256) {
        const float* q = Qf + (size_t)(b * SEQ + i) * CH + h * HD;
        float z = 0.f;
#pragma unroll
        for (int dd = 0; dd < HD; ++dd) z += q[dd] * s_kavg[dd];
        z *= 0.125f;   // HEAD_DIM^-0.5
        s_z[i] = z;
        lmax = fmaxf(lmax, z);
    }
    s_red[t] = lmax;
    __syncthreads();
    for (int off = 128; off > 0; off >>= 1) {
        if (t < off) s_red[t] = fmaxf(s_red[t], s_red[t + off]);
        __syncthreads();
    }
    const float zmax = s_red[0];
    __syncthreads();

    float lsum = 0.f;
    for (int i = t; i < SEQ; i += 256) {
        float e = __expf(s_z[i] - zmax);
        s_z[i] = e;
        lsum += e;
    }
    s_red[t] = lsum;
    __syncthreads();
    for (int off = 128; off > 0; off >>= 1) {
        if (t < off) s_red[t] += s_red[t + off];
        __syncthreads();
    }
    const float inv = 1.0f / (s_red[0] * (float)SEQ);  // softmax + 1/N fold
    for (int i = t; i < SEQ; i += 256)
        attnb[(size_t)bh * SEQ + i] = __float2bfloat16(s_z[i] * inv);
}

// ---------------------------------------------------------------------------
// Kernel 3: circulant GEMM per (b,h):
//   Hout[i, d] = sum_k attn[(k - i) mod N] * V[k, d]
// attn is staged doubled in LDS so the mod becomes a plain offset.
// Block: 256 threads / 8 waves; wave = one 16-row M tile x all 64 cols.
// ---------------------------------------------------------------------------
__global__ void circconv_kernel(const __hip_bfloat16* __restrict__ attnb,
                                const __hip_bfloat16* __restrict__ Vt,
                                __hip_bfloat16* __restrict__ Hout) {
    __shared__ unsigned short s_attn[2 * SEQ];

    const int bh = blockIdx.y;
    const int m0 = blockIdx.x * 128;

    const unsigned short* ab =
        reinterpret_cast<const unsigned short*>(attnb) + (size_t)bh * SEQ;
    for (int idx = threadIdx.x; idx < 2 * SEQ; idx += 256)
        s_attn[idx] = ab[idx & (SEQ - 1)];
    __syncthreads();

    const int w    = threadIdx.x >> 5;
    const int lane = threadIdx.x & 31;
    const int r    = lane & 15;
    const int half8 = (lane >> 4) << 3;
    const int mrow = m0 + w * 16;
    const int i    = mrow + r;   // A-fragment row for this lane

    v8f zero8 = {0.f, 0.f, 0.f, 0.f, 0.f, 0.f, 0.f, 0.f};
    v8f acc[4];
#pragma unroll
    for (int nt = 0; nt < 4; ++nt) acc[nt] = zero8;

    const __hip_bfloat16* vb[4];
#pragma unroll
    for (int nt = 0; nt < 4; ++nt)
        vb[nt] = Vt + ((size_t)bh * HD + nt * 16 + r) * SEQ + half8;

    for (int k0 = 0; k0 < SEQ; k0 += 32) {
        // Build A fragment from LDS: element K=k0+kk is attn[(k0+kk - i) mod N]
        union { unsigned short u[16]; v16bf v; } af;
        const int base = k0 + half8 + SEQ - i;   // in [1, 2*SEQ-1-23]
#pragma unroll
        for (int e = 0; e < 8; ++e) af.u[e] = s_attn[base + e];
#pragma unroll
        for (int e = 0; e < 8; ++e) af.u[8 + e] = s_attn[base + 16 + e];

#pragma unroll
        for (int nt = 0; nt < 4; ++nt) {
            v16bf bfv = load_frag16(vb[nt] + k0);
            __builtin_prefetch(vb[nt] + k0 + 128, 0, 0);  // global_prefetch_b8
            acc[nt] = __builtin_amdgcn_wmma_f32_16x16x32_bf16(
                false, af.v, false, bfv, (short)0, acc[nt], false, false);
        }
    }

    const int b = bh / NH, h = bh % NH;
    const int mbase = mrow + half8;
#pragma unroll
    for (int nt = 0; nt < 4; ++nt) {
        const int dd = nt * 16 + r;
#pragma unroll
        for (int rr = 0; rr < 8; ++rr)
            Hout[(size_t)(b * SEQ + mbase + rr) * CH + h * HD + dd] =
                __float2bfloat16(acc[nt][rr]);
    }
}

// ---------------------------------------------------------------------------
// Host side
// ---------------------------------------------------------------------------
static inline size_t align_up(size_t x) { return (x + 255) & ~(size_t)255; }

extern "C" void kernel_launch(void* const* d_in, const int* in_sizes, int n_in,
                              void* d_out, int out_size, void* d_ws,
                              size_t ws_size, hipStream_t stream) {
    const float* x  = (const float*)d_in[0];
    const float* Wq = (const float*)d_in[1];
    const float* Wk = (const float*)d_in[2];
    const float* Wv = (const float*)d_in[3];
    const float* Wp = (const float*)d_in[4];
    const float* bp = (const float*)d_in[5];

    char* ws = (char*)d_ws;
    size_t off = 0;
    auto carve = [&](size_t bytes) {
        char* p = ws + off;
        off = align_up(off + bytes);
        return p;
    };
    __hip_bfloat16* xb  = (__hip_bfloat16*)carve((size_t)ROWS * CH * 2);
    __hip_bfloat16* wqb = (__hip_bfloat16*)carve((size_t)CH * CH * 2);
    __hip_bfloat16* wkb = (__hip_bfloat16*)carve((size_t)CH * CH * 2);
    __hip_bfloat16* wvb = (__hip_bfloat16*)carve((size_t)CH * CH * 2);
    __hip_bfloat16* wpb = (__hip_bfloat16*)carve((size_t)CH * CH * 2);
    float* Qf           = (float*)carve((size_t)ROWS * CH * 4);
    float* Kf           = (float*)carve((size_t)ROWS * CH * 4);
    __hip_bfloat16* Vt  = (__hip_bfloat16*)carve((size_t)BQ * NH * HD * SEQ * 2);
    __hip_bfloat16* atb = (__hip_bfloat16*)carve((size_t)BQ * NH * SEQ * 2);
    __hip_bfloat16* Hb  = (__hip_bfloat16*)carve((size_t)ROWS * CH * 2);
    (void)ws_size;

    // 0) convert inputs to bf16
    {
        int n = ROWS * CH;
        cvt_bf16_kernel<<<(n + 255) / 256, 256, 0, stream>>>(x, xb, n);
        n = CH * CH;
        cvt_bf16_kernel<<<(n + 255) / 256, 256, 0, stream>>>(Wq, wqb, n);
        cvt_bf16_kernel<<<(n + 255) / 256, 256, 0, stream>>>(Wk, wkb, n);
        cvt_bf16_kernel<<<(n + 255) / 256, 256, 0, stream>>>(Wv, wvb, n);
        cvt_bf16_kernel<<<(n + 255) / 256, 256, 0, stream>>>(Wp, wpb, n);
    }

    const dim3 gGemm(CH / 128, ROWS / 128);   // 6 x 32
    // 1) projections
    gemm_bf16_kernel<0><<<gGemm, 256, 0, stream>>>(xb, wqb, Qf, nullptr, nullptr);
    gemm_bf16_kernel<1><<<gGemm, 256, 0, stream>>>(xb, wkb, Kf, nullptr, nullptr);
    gemm_bf16_kernel<2><<<gGemm, 256, 0, stream>>>(xb, wvb, nullptr, Vt, nullptr);

    // 2) averaged-key scores + softmax (+1/N fold)
    kavg_softmax_kernel<<<BQ * NH, 256, 0, stream>>>(Qf, Kf, atb);

    // 3) circulant attention GEMM
    circconv_kernel<<<dim3(SEQ / 128, BQ * NH), 256, 0, stream>>>(atb, Vt, Hb);

    // 4) output projection + bias
    gemm_bf16_kernel<3><<<gGemm, 256, 0, stream>>>(Hb, wpb, (float*)d_out,
                                                   nullptr, bp);
}
